// RNN_3255585210799
// MI455X (gfx1250) — compile-verified, matrix operand order checked by reading, not run on hardware
//
#include <hip/hip_runtime.h>
#include <hip/hip_bf16.h>
#include <math.h>

typedef __attribute__((ext_vector_type(16))) _Float16 v16h;
typedef __attribute__((ext_vector_type(8)))  float    v8f;

// Payload type for the async b128 LDS-load builtin (matches the diagnostic:
// "__vector_size__(4*sizeof(int)) int" in AS1 / AS3).
typedef int v4i_vs __attribute__((vector_size(16)));
typedef __attribute__((address_space(1))) v4i_vs* as1_v4i;
typedef __attribute__((address_space(3))) v4i_vs* as3_v4i;

#define T_STEPS  1024
#define BATCH    256
#define IN       128
#define HID      64
#define G4       256   // 4*H
#define MB       16    // batch rows per block
#define NTHREADS 256   // 8 waves

#if defined(__gfx1250__) && \
    __has_builtin(__builtin_amdgcn_global_load_async_to_lds_b128) && \
    __has_builtin(__builtin_amdgcn_s_wait_asynccnt)
#define USE_ASYNC 1
#else
#define USE_ASYNC 0
#endif

// K index held in half-slot h (0..15) for lane-group g (lane>>4), per the
// 16-bit 16x32 A-matrix layout (ISA 7.12.2). B mirrors it with lane = N.
__device__ __forceinline__ int kmap(int h, int g) {
  return (h < 8) ? (g * 8 + h) : (16 + g * 8 + (h - 8));
}

__device__ __forceinline__ v8f wmma_f16(v16h a, v16h b, v8f c) {
  return __builtin_amdgcn_wmma_f32_16x16x32_f16(false, a, false, b,
                                                (short)0, c, false, false);
}

__device__ __forceinline__ float sigmoidf_(float x) {
  return 1.0f / (1.0f + __expf(-x));
}

#if USE_ASYNC
#define PREFETCH_X(buf, tt)                                                    \
  {                                                                            \
    _Pragma("unroll")                                                          \
    for (int q_ = 0; q_ < 2; ++q_) {                                           \
      int flat_ = tid + q_ * NTHREADS; /* 0..511 float4 tiles */               \
      int row_  = flat_ >> 5;                                                  \
      int c4_   = (flat_ & 31) * 4;                                            \
      const float* gp_ = x + ((size_t)(tt)*BATCH + (b0 + row_)) * IN + c4_;    \
      __builtin_amdgcn_global_load_async_to_lds_b128(                          \
          (as1_v4i)(__attribute__((address_space(1))) void*)                   \
              const_cast<float*>(gp_),                                         \
          (as3_v4i)(__attribute__((address_space(3))) void*)                   \
              (&sX[buf][row_][c4_]),                                           \
          0, 0);                                                               \
    }                                                                          \
  }
#endif

// ---------------------------------------------------------------------------
// Kernel 1: forward LSTM scan. Each block owns 16 batch rows for all 1024
// steps. W_ih^T / W_hh^T live in registers as f16 WMMA B-fragments; h in LDS
// (f16); cell state c in registers (fixed thread ownership). Per step:
// gates[16,256] = x_t@W_ih^T + h@W_hh^T + b  via 12 WMMAs/wave, then
// activations + cell update. Next x_t slice streamed via async LDS loads.
// ---------------------------------------------------------------------------
__global__ void __launch_bounds__(NTHREADS)
lstm_fwd_scan(const float* __restrict__ x, const float* __restrict__ Wih,
              const float* __restrict__ Whh, const float* __restrict__ bias,
              float* __restrict__ hfwd) {
  __shared__ float    sX[2][MB][IN];   // 16 KB double-buffered x_t slice
  __shared__ _Float16 sH[MB][HID];     // 2 KB  hidden state (A operand src)
  __shared__ float    sG[MB][G4];      // 16 KB activated gates

  const int tid  = threadIdx.x;
  const int wave = tid >> 5;
  const int lane = tid & 31;
  const int lm   = lane & 15;
  const int lg   = lane >> 4;
  const int b0   = blockIdx.x * MB;

  const int n0 = (wave * 2 + 0) * 16 + lm;  // N column of tile 0 for this lane
  const int n1 = (wave * 2 + 1) * 16 + lm;  // N column of tile 1

  // Constant B-operand fragments (W^T), f16, kept in VGPRs for whole scan.
  v16h bWih[2][4], bWhh[2][2];
#pragma unroll
  for (int t2 = 0; t2 < 2; ++t2) {
    const int n = t2 ? n1 : n0;
#pragma unroll
    for (int kc = 0; kc < 4; ++kc)
#pragma unroll
      for (int h = 0; h < 16; ++h)
        bWih[t2][kc][h] = (_Float16)Wih[n * IN + kc * 32 + kmap(h, lg)];
#pragma unroll
    for (int kc = 0; kc < 2; ++kc)
#pragma unroll
      for (int h = 0; h < 16; ++h)
        bWhh[t2][kc][h] = (_Float16)Whh[n * HID + kc * 32 + kmap(h, lg)];
  }
  const float bias0 = bias[n0];
  const float bias1 = bias[n1];

  for (int i = tid; i < MB * HID; i += NTHREADS)
    ((_Float16*)sH)[i] = (_Float16)0.0f;

  float cst[4] = {0.f, 0.f, 0.f, 0.f};  // cell state, fixed (b,j) per slot

#if USE_ASYNC
  PREFETCH_X(0, 0);
#endif

  for (int t = 0; t < T_STEPS; ++t) {
#if USE_ASYNC
    const int cur = t & 1;
    __builtin_amdgcn_s_wait_asynccnt(0);  // my step-t loads landed
    __syncthreads();                      // everyone's landed; sH(t-1) visible
    {
      const int tn = (t + 1 < T_STEPS) ? (t + 1) : (T_STEPS - 1);
      PREFETCH_X(cur ^ 1, tn);            // overlaps with this step's compute
    }
#else
    const int cur = 0;
    __syncthreads();
#pragma unroll
    for (int q = 0; q < 2; ++q) {
      int flat = tid + q * NTHREADS;
      int row  = flat >> 5;
      int c4   = (flat & 31) * 4;
      const float4 v =
          *(const float4*)(x + ((size_t)t * BATCH + (b0 + row)) * IN + c4);
      *(float4*)(&sX[0][row][c4]) = v;
    }
    __syncthreads();
#endif

    v8f c0, c1;
#pragma unroll
    for (int r = 0; r < 8; ++r) { c0[r] = bias0; c1[r] = bias1; }

    // x_t @ W_ih^T  (K = 128 -> 4 WMMA k-steps)
#pragma unroll
    for (int kc = 0; kc < 4; ++kc) {
      v16h a;
#pragma unroll
      for (int h = 0; h < 16; ++h)
        a[h] = (_Float16)sX[cur][lm][kc * 32 + kmap(h, lg)];
      c0 = wmma_f16(a, bWih[0][kc], c0);
      c1 = wmma_f16(a, bWih[1][kc], c1);
    }
    // h @ W_hh^T  (K = 64 -> 2 WMMA k-steps)
#pragma unroll
    for (int kc = 0; kc < 2; ++kc) {
      v16h a;
#pragma unroll
      for (int h = 0; h < 16; ++h)
        a[h] = sH[lm][kc * 32 + kmap(h, lg)];
      c0 = wmma_f16(a, bWhh[0][kc], c0);
      c1 = wmma_f16(a, bWhh[1][kc], c1);
    }

    // Activate and scatter gates to LDS. gate: 0=i,1=f,2=g,3=o by column/64.
#pragma unroll
    for (int t2 = 0; t2 < 2; ++t2) {
      const int n    = t2 ? n1 : n0;
      const int gate = n >> 6;
      const v8f c    = t2 ? c1 : c0;
#pragma unroll
      for (int r = 0; r < 8; ++r) {
        const int m = lg * 8 + r;   // C/D layout: M = r + 8*(lane>=16)
        float v = c[r];
        v = (gate == 2) ? tanhf(v) : sigmoidf_(v);
        sG[m][n] = v;
      }
    }
    __syncthreads();

    // Cell update: each thread owns 4 fixed (b,j); c never leaves registers.
#pragma unroll
    for (int r = 0; r < 4; ++r) {
      const int flat = tid + NTHREADS * r;
      const int b = flat >> 6;
      const int j = flat & 63;
      const float ig = sG[b][j];
      const float fg = sG[b][64 + j];
      const float gg = sG[b][128 + j];
      const float og = sG[b][192 + j];
      cst[r] = fg * cst[r] + ig * gg;
      const float hv = og * tanhf(cst[r]);
      sH[b][j] = (_Float16)hv;
      if (t == T_STEPS - 1) hfwd[(size_t)(b0 + b) * HID + j] = hv;
    }
  }
}

// ---------------------------------------------------------------------------
// Kernel 2: reverse-direction cell on x[T-1] with zero state:
// gates = x_last @ W_ih_r^T + b_r ; c = i*g ; h = o*tanh(c).
// ---------------------------------------------------------------------------
__global__ void __launch_bounds__(NTHREADS)
lstm_rev_last(const float* __restrict__ x, const float* __restrict__ Wih,
              const float* __restrict__ bias, float* __restrict__ hbwd) {
  __shared__ float sG[MB][G4];

  const int tid  = threadIdx.x;
  const int wave = tid >> 5;
  const int lane = tid & 31;
  const int lm   = lane & 15;
  const int lg   = lane >> 4;
  const int b0   = blockIdx.x * MB;
  const int n0   = (wave * 2 + 0) * 16 + lm;
  const int n1   = (wave * 2 + 1) * 16 + lm;

  v8f c0, c1;
#pragma unroll
  for (int r = 0; r < 8; ++r) { c0[r] = bias[n0]; c1[r] = bias[n1]; }

  const float* xl = x + (size_t)(T_STEPS - 1) * BATCH * IN + (size_t)b0 * IN;
#pragma unroll
  for (int kc = 0; kc < 4; ++kc) {
    v16h a, w0, w1;
#pragma unroll
    for (int h = 0; h < 16; ++h) {
      const int k = kc * 32 + kmap(h, lg);
      a[h]  = (_Float16)xl[lm * IN + k];
      w0[h] = (_Float16)Wih[n0 * IN + k];
      w1[h] = (_Float16)Wih[n1 * IN + k];
    }
    c0 = wmma_f16(a, w0, c0);
    c1 = wmma_f16(a, w1, c1);
  }

#pragma unroll
  for (int t2 = 0; t2 < 2; ++t2) {
    const int n    = t2 ? n1 : n0;
    const int gate = n >> 6;
    const v8f c    = t2 ? c1 : c0;
#pragma unroll
    for (int r = 0; r < 8; ++r) {
      const int m = lg * 8 + r;
      float v = c[r];
      v = (gate == 2) ? tanhf(v) : sigmoidf_(v);
      sG[m][n] = v;
    }
  }
  __syncthreads();

#pragma unroll
  for (int r = 0; r < 4; ++r) {
    const int flat = tid + NTHREADS * r;
    const int b = flat >> 6;
    const int j = flat & 63;
    const float cc = sG[b][j] * sG[b][128 + j];      // i*g (f*c0 == 0)
    const float hv = sG[b][192 + j] * tanhf(cc);     // o*tanh(c)
    hbwd[(size_t)(b0 + b) * HID + j] = hv;
  }
}

// ---------------------------------------------------------------------------
// Kernel 3: MLP head. h_cat[256,128] -> fc1(64,relu) -> fc2(32,relu) -> fc3(10)
// fc1/fc2 via WMMA, fc3 scalar (N=10). Single workgroup, 8 waves.
// ---------------------------------------------------------------------------
__global__ void __launch_bounds__(NTHREADS)
mlp_head(const float* __restrict__ hfwd, const float* __restrict__ hbwd,
         const float* __restrict__ fc1w, const float* __restrict__ fc1b,
         const float* __restrict__ fc2w, const float* __restrict__ fc2b,
         const float* __restrict__ fc3w, const float* __restrict__ fc3b,
         float* __restrict__ out) {
  __shared__ _Float16 sH1[BATCH][64];  // 32 KB
  __shared__ _Float16 sH2[BATCH][32];  // 16 KB

  const int tid  = threadIdx.x;
  const int wave = tid >> 5;
  const int lane = tid & 31;
  const int lm   = lane & 15;
  const int lg   = lane >> 4;

  // fc1: [256,128] @ [128,64] ; each wave: 2 M-tiles x 4 N-tiles
#pragma unroll
  for (int mt = 0; mt < 2; ++mt) {
    const int m0  = (wave * 2 + mt) * 16;
    const int row = m0 + lm;
    v8f acc[4];
#pragma unroll
    for (int nt = 0; nt < 4; ++nt)
#pragma unroll
      for (int r = 0; r < 8; ++r) acc[nt][r] = fc1b[nt * 16 + lm];
#pragma unroll
    for (int kc = 0; kc < 4; ++kc) {
      v16h a;
#pragma unroll
      for (int h = 0; h < 16; ++h) {
        const int k = kc * 32 + kmap(h, lg);
        a[h] = (_Float16)((k < HID) ? hfwd[(size_t)row * HID + k]
                                    : hbwd[(size_t)row * HID + (k - HID)]);
      }
#pragma unroll
      for (int nt = 0; nt < 4; ++nt) {
        v16h bfrag;
#pragma unroll
        for (int h = 0; h < 16; ++h) {
          const int k = kc * 32 + kmap(h, lg);
          bfrag[h] = (_Float16)fc1w[(nt * 16 + lm) * 128 + k];
        }
        acc[nt] = wmma_f16(a, bfrag, acc[nt]);
      }
    }
#pragma unroll
    for (int nt = 0; nt < 4; ++nt)
#pragma unroll
      for (int r = 0; r < 8; ++r) {
        float v = acc[nt][r];
        sH1[m0 + lg * 8 + r][nt * 16 + lm] = (_Float16)(v > 0.f ? v : 0.f);
      }
  }
  __syncthreads();

  // fc2: [256,64] @ [64,32]
#pragma unroll
  for (int mt = 0; mt < 2; ++mt) {
    const int m0 = (wave * 2 + mt) * 16;
    v8f acc[2];
#pragma unroll
    for (int nt = 0; nt < 2; ++nt)
#pragma unroll
      for (int r = 0; r < 8; ++r) acc[nt][r] = fc2b[nt * 16 + lm];
#pragma unroll
    for (int kc = 0; kc < 2; ++kc) {
      v16h a;
#pragma unroll
      for (int h = 0; h < 16; ++h)
        a[h] = sH1[m0 + lm][kc * 32 + kmap(h, lg)];
#pragma unroll
      for (int nt = 0; nt < 2; ++nt) {
        v16h bfrag;
#pragma unroll
        for (int h = 0; h < 16; ++h) {
          const int k = kc * 32 + kmap(h, lg);
          bfrag[h] = (_Float16)fc2w[(nt * 16 + lm) * 64 + k];
        }
        acc[nt] = wmma_f16(a, bfrag, acc[nt]);
      }
    }
#pragma unroll
    for (int nt = 0; nt < 2; ++nt)
#pragma unroll
      for (int r = 0; r < 8; ++r) {
        float v = acc[nt][r];
        sH2[m0 + lg * 8 + r][nt * 16 + lm] = (_Float16)(v > 0.f ? v : 0.f);
      }
  }
  __syncthreads();

  // fc3: [256,32] @ [32,10], one row per thread
  {
    const int b = tid;
#pragma unroll
    for (int u = 0; u < 10; ++u) {
      float s = fc3b[u];
#pragma unroll
      for (int k = 0; k < 32; ++k) s += (float)sH2[b][k] * fc3w[u * 32 + k];
      out[b * 10 + u] = s;
    }
  }
}

extern "C" void kernel_launch(void* const* d_in, const int* in_sizes, int n_in,
                              void* d_out, int out_size, void* d_ws,
                              size_t ws_size, hipStream_t stream) {
  (void)in_sizes; (void)n_in; (void)out_size; (void)ws_size;
  const float* x     = (const float*)d_in[0];
  const float* Wih_f = (const float*)d_in[1];
  const float* Whh_f = (const float*)d_in[2];
  const float* b_f   = (const float*)d_in[3];
  const float* Wih_r = (const float*)d_in[4];
  // d_in[5] = W_hh_r: unused (h0 == 0 makes its contribution vanish)
  const float* b_r   = (const float*)d_in[6];
  const float* fc1w  = (const float*)d_in[7];
  const float* fc1b  = (const float*)d_in[8];
  const float* fc2w  = (const float*)d_in[9];
  const float* fc2b  = (const float*)d_in[10];
  const float* fc3w  = (const float*)d_in[11];
  const float* fc3b  = (const float*)d_in[12];
  float* out  = (float*)d_out;
  float* hfwd = (float*)d_ws;                 // [256,64] f32
  float* hbwd = hfwd + BATCH * HID;           // [256,64] f32

  lstm_fwd_scan<<<dim3(BATCH / MB), dim3(NTHREADS), 0, stream>>>(
      x, Wih_f, Whh_f, b_f, hfwd);
  lstm_rev_last<<<dim3(BATCH / MB), dim3(NTHREADS), 0, stream>>>(
      x, Wih_r, b_r, hbwd);
  mlp_head<<<dim3(1), dim3(NTHREADS), 0, stream>>>(
      hfwd, hbwd, fc1w, fc1b, fc2w, fc2b, fc3w, fc3b, out);
}
